// DifferentialAttention_19602230739616
// MI455X (gfx1250) — compile-verified, matrix operand order checked by reading, not run on hardware
//
#include <hip/hip_runtime.h>

#define S_ 4096
#define DIM_ 1024
#define H_ 8
#define HD_ 64
#define P_ 1024   // 2*H*HD
#define MW_ (S_ / 32)  // mask words per row

typedef __attribute__((ext_vector_type(16))) __bf16 v16bf;
typedef __attribute__((ext_vector_type(8)))  __bf16 v8bf;
typedef __attribute__((ext_vector_type(8)))  float  v8f;
typedef __attribute__((ext_vector_type(4)))  float  v4f;

// ---------------------------------------------------------------------------
// WMMA helper: D = A(16x32 bf16) * B(32x16 bf16) + C(16x16 f32)
// ---------------------------------------------------------------------------
static __device__ __forceinline__ v8f wmma_bf(v16bf a, v16bf b, v8f c) {
  return __builtin_amdgcn_wmma_f32_16x16x32_bf16(
      /*neg_a=*/false, a, /*neg_b=*/false, b,
      /*c_mod=*/(short)0, c, /*reuse_a=*/false, /*reuse_b=*/false);
}

// ---------------------------------------------------------------------------
// Fragment loader (A-layout, 16x32 bf16 tile from row-major memory).
// The CDNA5 B-fragment of B (32x16) equals the A-fragment of B^T, so this
// loader serves both operands when the B matrix is stored transposed.
// ---------------------------------------------------------------------------
static __device__ __forceinline__ v16bf load_frag(const __bf16* __restrict__ base,
                                                  int stride) {
  const int lane = threadIdx.x & 31;
  const int r    = lane & 15;
  const int kb   = (lane & 16) ? 8 : 0;
  const __bf16* p = base + (size_t)r * stride + kb;
  v8bf lo = *(const v8bf*)(p);
  v8bf hi = *(const v8bf*)(p + 16);
  v16bf f;
#pragma unroll
  for (int i = 0; i < 8; ++i) { f[i] = lo[i]; f[8 + i] = hi[i]; }
  return f;
}

// ---------------------------------------------------------------------------
// Conversions
// ---------------------------------------------------------------------------
__global__ void cvt_bf16_kernel(const float* __restrict__ in,
                                __bf16* __restrict__ out, int n) {
  int i = blockIdx.x * blockDim.x + threadIdx.x;
  if (i < n) out[i] = (__bf16)in[i];
}

// in: (1024,1024) row-major (k,n) -> out: (n,k) bf16
__global__ void transpose_cvt_kernel(const float* __restrict__ in,
                                     __bf16* __restrict__ out) {
  int i = blockIdx.x * blockDim.x + threadIdx.x;  // 0 .. 2^20-1
  int k = i >> 10, n = i & 1023;
  out[((size_t)n << 10) + k] = (__bf16)in[i];
}

// ---------------------------------------------------------------------------
// Pack boolean mask bytes into u32 bitmask words: bits[s][w] bit j = mask[s][32w+j]
// ---------------------------------------------------------------------------
static __device__ __forceinline__ unsigned int pack4(unsigned int v, int base) {
  unsigned int b = 0;
#pragma unroll
  for (int j = 0; j < 4; ++j)
    if ((v >> (8 * j)) & 0xFFu) b |= 1u << (base + j);
  return b;
}

__global__ void mask_bits_kernel(const unsigned char* __restrict__ mask,
                                 unsigned int* __restrict__ bits) {
  int w = blockIdx.x * blockDim.x + threadIdx.x;  // 0 .. S_*MW_-1
  const uint4* q = (const uint4*)(mask + (size_t)w * 32);
  uint4 v0 = q[0], v1 = q[1];
  unsigned int b = 0;
  b |= pack4(v0.x, 0);  b |= pack4(v0.y, 4);
  b |= pack4(v0.z, 8);  b |= pack4(v0.w, 12);
  b |= pack4(v1.x, 16); b |= pack4(v1.y, 20);
  b |= pack4(v1.z, 24); b |= pack4(v1.w, 28);
  bits[w] = b;
}

// ---------------------------------------------------------------------------
// lambda_full = exp(sum(lq1*lk1)) - exp(sum(lq2*lk2)) + 0.2
// ---------------------------------------------------------------------------
__global__ void lambda_kernel(const float* __restrict__ q1, const float* __restrict__ k1,
                              const float* __restrict__ q2, const float* __restrict__ k2,
                              float* __restrict__ lam) {
  int i = threadIdx.x;  // 0..31
  float s1 = q1[i] * k1[i] + q1[i + 32] * k1[i + 32];
  float s2 = q2[i] * k2[i] + q2[i + 32] * k2[i + 32];
#pragma unroll
  for (int sh = 16; sh >= 1; sh >>= 1) {
    s1 += __shfl_xor(s1, sh, 32);
    s2 += __shfl_xor(s2, sh, 32);
  }
  if (i == 0) lam[0] = __expf(s1) - __expf(s2) + 0.2f;
}

// ---------------------------------------------------------------------------
// Pipelined GEMM: per wave 32x64 tile of C = A(bf16, row-major) * B,
// with B supplied transposed (N x K, row-major).  K = 1024.
// MODE 0: store bf16 row-major   (Q, K projections)
// MODE 1: store bf16 transposed  (V^T, P_ x S_)
// MODE 2: store f32  row-major   (final output)
// grid (S/32, N/64), block 32 (one wave)
// ---------------------------------------------------------------------------
template <int MODE>
__global__ void __launch_bounds__(32)
gemm_bf16_kernel(const __bf16* __restrict__ A,
                 const __bf16* __restrict__ BT,
                 void* __restrict__ outp) {
  const int s0   = blockIdx.x * 32;
  const int n0   = blockIdx.y * 64;
  const int lane = threadIdx.x;
  const int col  = lane & 15;
  const int rb   = (lane & 16) ? 8 : 0;

  const __bf16* arow0 = A + (size_t)s0 * DIM_;
  const __bf16* arow1 = arow0 + (size_t)16 * DIM_;
  const __bf16* brow0 = BT + (size_t)(n0 +  0) * DIM_;
  const __bf16* brow1 = BT + (size_t)(n0 + 16) * DIM_;
  const __bf16* brow2 = BT + (size_t)(n0 + 32) * DIM_;
  const __bf16* brow3 = BT + (size_t)(n0 + 48) * DIM_;

  v8f acc[2][4] = {};

  // prologue loads (kc = 0)
  v16bf a0 = load_frag(arow0, DIM_);
  v16bf a1 = load_frag(arow1, DIM_);
  v16bf b0 = load_frag(brow0, DIM_);
  v16bf b1 = load_frag(brow1, DIM_);
  v16bf b2 = load_frag(brow2, DIM_);
  v16bf b3 = load_frag(brow3, DIM_);

  for (int kc = 0; kc < DIM_; kc += 32) {
    const int kn = (kc + 32 < DIM_) ? kc + 32 : 0;  // dead prefetch on last iter
    // issue next-iteration loads BEFORE consuming current fragments so the
    // WMMAs only need a partial LOADcnt wait (overlap load latency).
    v16bf na0 = load_frag(arow0 + kn, DIM_);
    v16bf na1 = load_frag(arow1 + kn, DIM_);
    v16bf nb0 = load_frag(brow0 + kn, DIM_);
    v16bf nb1 = load_frag(brow1 + kn, DIM_);
    v16bf nb2 = load_frag(brow2 + kn, DIM_);
    v16bf nb3 = load_frag(brow3 + kn, DIM_);

    acc[0][0] = wmma_bf(a0, b0, acc[0][0]);
    acc[0][1] = wmma_bf(a0, b1, acc[0][1]);
    acc[0][2] = wmma_bf(a0, b2, acc[0][2]);
    acc[0][3] = wmma_bf(a0, b3, acc[0][3]);
    acc[1][0] = wmma_bf(a1, b0, acc[1][0]);
    acc[1][1] = wmma_bf(a1, b1, acc[1][1]);
    acc[1][2] = wmma_bf(a1, b2, acc[1][2]);
    acc[1][3] = wmma_bf(a1, b3, acc[1][3]);

    a0 = na0; a1 = na1;
    b0 = nb0; b1 = nb1; b2 = nb2; b3 = nb3;
  }

#pragma unroll
  for (int i = 0; i < 2; ++i) {
#pragma unroll
    for (int j = 0; j < 4; ++j) {
      const int n = n0 + 16 * j + col;
      if constexpr (MODE == 0) {
        __bf16* out = (__bf16*)outp;
#pragma unroll
        for (int r = 0; r < 8; ++r)
          out[(size_t)(s0 + 16 * i + rb + r) * P_ + n] = (__bf16)acc[i][j][r];
      } else if constexpr (MODE == 1) {
        __bf16* out = (__bf16*)outp;
        v8bf pk;
#pragma unroll
        for (int r = 0; r < 8; ++r) pk[r] = (__bf16)acc[i][j][r];
        *(v8bf*)(out + (size_t)n * S_ + s0 + 16 * i + rb) = pk;
      } else {
        float* out = (float*)outp;
#pragma unroll
        for (int r = 0; r < 8; ++r)
          out[(size_t)(s0 + 16 * i + rb + r) * DIM_ + n] = acc[i][j][r];
      }
    }
  }
}

// ---------------------------------------------------------------------------
// Flash attention with TRANSPOSED score tiles: scores^T = K_tile x Q^T, so the
// softmax key-reduction is in-lane (tree over accumulator regs) + ONE
// shfl_xor(16) per chunk instead of 64 ds_bpermutes.
// grid (S/16, H, 2), block 32 (one wave).
// Q,K row-major bf16 (S x 1024); Vt = V^T bf16 (1024 x S); out fp32 SxHx128.
// ---------------------------------------------------------------------------
__global__ void __launch_bounds__(32)
attn_kernel(const __bf16* __restrict__ Qb,
            const __bf16* __restrict__ Kb,
            const __bf16* __restrict__ Vt,
            const unsigned int* __restrict__ mbits,
            float* __restrict__ attn1, float* __restrict__ attn2) {
  __shared__ __align__(16) __bf16 pbuf[16 * 32];

  const int s0   = blockIdx.x * 16;
  const int h    = blockIdx.y;
  const int c    = blockIdx.z;
  float* out     = c ? attn2 : attn1;
  const int lane = threadIdx.x;
  const int col  = lane & 15;          // this lane's query column
  const int rb   = (lane & 16) ? 8 : 0;

  // Q fragments: serve as the B operand of the transposed score WMMA
  const __bf16* qbase = Qb + (size_t)s0 * P_ + h * 128 + c * 64;
  v16bf qf0 = load_frag(qbase + 0, P_);
  v16bf qf1 = load_frag(qbase + 32, P_);

  const __bf16* kcol  = Kb + h * 128 + c * 64;
  const __bf16* vbase = Vt + (size_t)(h * 128) * S_;
  const unsigned int* mrow = mbits + (size_t)(s0 + col) * MW_;

  v8f O[8] = {};                        // O^T tiles: rows = value dims, cols = queries
  float m = -3.0e38f, l = 0.f;          // per-lane softmax stats (one query each)
  const float scale = 0.125f;           // 1/sqrt(64)

  for (int t0 = 0; t0 < S_; t0 += 32) {
    // ---- K fragments (A operand), then mask word (1 load/lane) ----
    const __bf16* kbp = kcol + (size_t)t0 * P_;
    v16bf kf0 = load_frag(kbp + 0, P_);
    v16bf kf1 = load_frag(kbp + 32, P_);
    v16bf kf2 = load_frag(kbp + (size_t)16 * P_ + 0, P_);
    v16bf kf3 = load_frag(kbp + (size_t)16 * P_ + 32, P_);
    const unsigned int mw = mrow[t0 >> 5];

    // ---- transposed scores: rows = keys (t), cols = queries (s) ----
    v8f sc0 = {}, sc1 = {};
    sc0 = wmma_bf(kf0, qf0, sc0);
    sc0 = wmma_bf(kf1, qf1, sc0);
    sc1 = wmma_bf(kf2, qf0, sc1);
    sc1 = wmma_bf(kf3, qf1, sc1);

    // ---- online softmax over keys (in-lane tree + one xor-16 shuffle) ----
    const unsigned int sel0 = mw >> rb;          // bits for t = rb+r
    const unsigned int sel1 = mw >> (16 + rb);   // bits for t = 16+rb+r
    float x0[8], x1[8];
    float mx = -3.0e38f;
#pragma unroll
    for (int r = 0; r < 8; ++r) {
      x0[r] = sc0[r] * scale + (((sel0 >> r) & 1u) ? 0.f : -3.0e38f);
      x1[r] = sc1[r] * scale + (((sel1 >> r) & 1u) ? 0.f : -3.0e38f);
      mx = fmaxf(mx, fmaxf(x0[r], x1[r]));
    }
    mx = fmaxf(mx, __shfl_xor(mx, 16, 32));      // combine the two key-halves
    const float mn = fmaxf(m, mx);
    float rs = 0.f;
    v8bf pk0, pk1;
#pragma unroll
    for (int r = 0; r < 8; ++r) {
      float p0 = __expf(x0[r] - mn);
      float p1 = __expf(x1[r] - mn);
      rs += p0 + p1;
      pk0[r] = (__bf16)p0;
      pk1[r] = (__bf16)p1;
    }
    rs += __shfl_xor(rs, 16, 32);
    const float f = __expf(m - mn);
    l = l * f + rs;
    m = mn;

    // rescale running output by per-lane factor
#pragma unroll
    for (int e = 0; e < 8; ++e)
#pragma unroll
      for (int r = 0; r < 8; ++r) O[e][r] *= f;

    // ---- stage P row-major (s-major) in LDS: 2 x ds_store_b128 per lane ----
    __syncthreads();                     // previous chunk's reads are done
    *(v8bf*)&pbuf[col * 32 + rb]      = pk0;   // t slots rb..rb+7
    *(v8bf*)&pbuf[col * 32 + 16 + rb] = pk1;   // t slots 16+rb..16+rb+7
    __syncthreads();

    // ---- read P as B-fragment (A-layout of row-major P) ----
    v16bf pf;
    {
      const __bf16* p = &pbuf[(lane & 15) * 32 + rb];
      v8bf lo = *(const v8bf*)(p);
      v8bf hi = *(const v8bf*)(p + 16);
#pragma unroll
      for (int i = 0; i < 8; ++i) { pf[i] = lo[i]; pf[8 + i] = hi[i]; }
    }

    // ---- PV (transposed): O^T += V^T x P', V fragments in two groups ----
    {
      v16bf vf0 = load_frag(vbase + (size_t)(16 * 0) * S_ + t0, S_);
      v16bf vf1 = load_frag(vbase + (size_t)(16 * 1) * S_ + t0, S_);
      v16bf vf2 = load_frag(vbase + (size_t)(16 * 2) * S_ + t0, S_);
      v16bf vf3 = load_frag(vbase + (size_t)(16 * 3) * S_ + t0, S_);
      O[0] = wmma_bf(vf0, pf, O[0]);
      O[1] = wmma_bf(vf1, pf, O[1]);
      O[2] = wmma_bf(vf2, pf, O[2]);
      O[3] = wmma_bf(vf3, pf, O[3]);
    }
    {
      v16bf vf4 = load_frag(vbase + (size_t)(16 * 4) * S_ + t0, S_);
      v16bf vf5 = load_frag(vbase + (size_t)(16 * 5) * S_ + t0, S_);
      v16bf vf6 = load_frag(vbase + (size_t)(16 * 6) * S_ + t0, S_);
      v16bf vf7 = load_frag(vbase + (size_t)(16 * 7) * S_ + t0, S_);
      O[4] = wmma_bf(vf4, pf, O[4]);
      O[5] = wmma_bf(vf5, pf, O[5]);
      O[6] = wmma_bf(vf6, pf, O[6]);
      O[7] = wmma_bf(vf7, pf, O[7]);
    }
  }

  // ---- finalize: O^T element (e = 16*et + rb + r, s = col) is contiguous
  //      along e per lane -> vectorized b128 stores ----
  const float inv = 1.0f / l;
  float* obase = out + (size_t)(s0 + col) * P_ + h * 128;
#pragma unroll
  for (int e = 0; e < 8; ++e) {
    v4f lo, hi;
#pragma unroll
    for (int r = 0; r < 4; ++r) { lo[r] = O[e][r] * inv; hi[r] = O[e][4 + r] * inv; }
    *(v4f*)(obase + 16 * e + rb)     = lo;
    *(v4f*)(obase + 16 * e + rb + 4) = hi;
  }
}

// ---------------------------------------------------------------------------
// attn = attn1 - lambda*attn2 ; RMS-norm over 128 ; * subln * (1-0.2) -> bf16
// grid (S*H), block 128
// ---------------------------------------------------------------------------
__global__ void combine_rms_kernel(const float* __restrict__ a1,
                                   const float* __restrict__ a2,
                                   const float* __restrict__ lam,
                                   const float* __restrict__ subln,
                                   __bf16* __restrict__ An) {
  const int sh_ = blockIdx.x;
  const int s = sh_ >> 3, h = sh_ & 7;
  const int e = threadIdx.x;  // 0..127
  const size_t idx = (size_t)s * P_ + h * 128 + e;
  float a = a1[idx] - lam[0] * a2[idx];
  float ss = a * a;
#pragma unroll
  for (int sh2 = 16; sh2 >= 1; sh2 >>= 1) ss += __shfl_xor(ss, sh2, 32);
  __shared__ float red[4];
  if ((threadIdx.x & 31) == 0) red[threadIdx.x >> 5] = ss;
  __syncthreads();
  float tot = red[0] + red[1] + red[2] + red[3];
  float rn = rsqrtf(tot * (1.0f / 128.0f) + 1e-5f);
  An[idx] = (__bf16)(a * rn * subln[e] * 0.8f);
}

// ---------------------------------------------------------------------------
// Workspace layout (bytes)
// ---------------------------------------------------------------------------
#define MB ((size_t)1 << 20)
#define OFF_XB    ((size_t)0)        // S*DIM bf16      = 8 MB
#define OFF_WQT   (8 * MB)           // 1024x1024 bf16  = 2 MB
#define OFF_WKT   (10 * MB)
#define OFF_WVT   (12 * MB)
#define OFF_WOT   (14 * MB)
#define OFF_QB    (16 * MB)          // S*P bf16 = 8 MB
#define OFF_KB    (24 * MB)
#define OFF_VT    (32 * MB)          // P*S bf16 = 8 MB (V^T)
#define OFF_A1    (40 * MB)          // S*P f32 = 16 MB
#define OFF_A2    (56 * MB)
#define OFF_AN    (72 * MB)          // S*P bf16 = 8 MB
#define OFF_LAM   (80 * MB)          // 4 B
#define OFF_MBITS (81 * MB)          // S*(S/32) u32 = 2 MB

extern "C" void kernel_launch(void* const* d_in, const int* in_sizes, int n_in,
                              void* d_out, int out_size, void* d_ws, size_t ws_size,
                              hipStream_t stream) {
  (void)in_sizes; (void)n_in; (void)out_size; (void)ws_size;

  const float* x   = (const float*)d_in[0];
  const unsigned char* mask = (const unsigned char*)d_in[1];
  const float* Wq  = (const float*)d_in[2];
  const float* Wk  = (const float*)d_in[3];
  const float* Wv  = (const float*)d_in[4];
  const float* Wo  = (const float*)d_in[5];
  const float* lq1 = (const float*)d_in[6];
  const float* lk1 = (const float*)d_in[7];
  const float* lq2 = (const float*)d_in[8];
  const float* lk2 = (const float*)d_in[9];
  const float* subln = (const float*)d_in[10];

  char* ws = (char*)d_ws;
  __bf16* xb  = (__bf16*)(ws + OFF_XB);
  __bf16* WqT = (__bf16*)(ws + OFF_WQT);
  __bf16* WkT = (__bf16*)(ws + OFF_WKT);
  __bf16* WvT = (__bf16*)(ws + OFF_WVT);
  __bf16* WoT = (__bf16*)(ws + OFF_WOT);
  __bf16* Qb  = (__bf16*)(ws + OFF_QB);
  __bf16* Kb  = (__bf16*)(ws + OFF_KB);
  __bf16* Vt  = (__bf16*)(ws + OFF_VT);
  float*  a1  = (float*)(ws + OFF_A1);
  float*  a2  = (float*)(ws + OFF_A2);
  __bf16* An  = (__bf16*)(ws + OFF_AN);
  float*  lam = (float*)(ws + OFF_LAM);
  unsigned int* mbits = (unsigned int*)(ws + OFF_MBITS);
  float*  out = (float*)d_out;

  // 1) conversions + mask packing
  {
    int n = S_ * DIM_;
    cvt_bf16_kernel<<<dim3((n + 255) / 256), dim3(256), 0, stream>>>(x, xb, n);
    transpose_cvt_kernel<<<dim3(4096), dim3(256), 0, stream>>>(Wq, WqT);
    transpose_cvt_kernel<<<dim3(4096), dim3(256), 0, stream>>>(Wk, WkT);
    transpose_cvt_kernel<<<dim3(4096), dim3(256), 0, stream>>>(Wv, WvT);
    transpose_cvt_kernel<<<dim3(4096), dim3(256), 0, stream>>>(Wo, WoT);
    mask_bits_kernel<<<dim3(S_ * MW_ / 256), dim3(256), 0, stream>>>(mask, mbits);
    lambda_kernel<<<dim3(1), dim3(32), 0, stream>>>(lq1, lk1, lq2, lk2, lam);
  }
  // 2) projections (Q, K row-major; V transposed)
  {
    dim3 g(S_ / 32, P_ / 64);
    gemm_bf16_kernel<0><<<g, dim3(32), 0, stream>>>(xb, WqT, Qb);
    gemm_bf16_kernel<0><<<g, dim3(32), 0, stream>>>(xb, WkT, Kb);
    gemm_bf16_kernel<1><<<g, dim3(32), 0, stream>>>(xb, WvT, Vt);
  }
  // 3) flash attention, both components
  attn_kernel<<<dim3(S_ / 16, H_, 2), dim3(32), 0, stream>>>(Qb, Kb, Vt, mbits, a1, a2);
  // 4) differential combine + RMS norm
  combine_rms_kernel<<<dim3(S_ * H_), dim3(128), 0, stream>>>(a1, a2, lam, subln, An);
  // 5) output projection
  gemm_bf16_kernel<2><<<dim3(S_ / 32, DIM_ / 64), dim3(32), 0, stream>>>(An, WoT, out);
}
